// EnhancedTransformerNetwork_20521353740569
// MI455X (gfx1250) — matrix operand order repaired and checked by reading, not compiled, not run on hardware
//
#include <hip/hip_runtime.h>
#include <hip/hip_bf16.h>

// ---------------------------------------------------------------------------
// EnhancedTransformerNetwork for MI455X (gfx1250), bf16 WMMA implementation.
//
// The two K=512 GEMM families (projections, attn@V) stage the block-shared
// 16xK A-strip into LDS with the Tensor Data Mover (tensor_load_to_lds +
// s_wait_tensorcnt), removing the 8x duplicated A reads across the 8 waves
// of a block. A-fragments then come from LDS (ds_load_b128), B-fragments
// stay as K-contiguous global_load_b128.
// ---------------------------------------------------------------------------

typedef __bf16 bf16_t;
typedef __bf16 bf16x16 __attribute__((ext_vector_type(16)));
typedef float  f32x8   __attribute__((ext_vector_type(8)));
typedef unsigned int u32x4 __attribute__((ext_vector_type(4)));
typedef int          i32x8 __attribute__((ext_vector_type(8)));
typedef int          i32x4 __attribute__((ext_vector_type(4)));

struct alignas(16) Bytes16 { unsigned long long a, b; };

union Frag {
    bf16x16 v;
    Bytes16 q[2];
};

// ---------------------------------------------------------------------------
// Tensor Data Mover: 2D tile load (rows x k_elems of bf16) global -> LDS.
// D# per CDNA5 ISA ch.8: group0 = {count=1, lds_addr, global_addr, type=2},
// group1 = {data_size=2B, tensor_dim0=k, tensor_dim1=rows, tile_dim0=k,
//           tile_dim1=rows, tensor_dim0_stride=ld}. Groups 2/3 zero (2D).
// This toolchain exposes the 6-arg builtin:
//   (u32x4 g0, i32x8 g1, i32x4, i32x4, i32x8, i32 cpol)
// ---------------------------------------------------------------------------
__device__ __forceinline__ void tdm_load_2d(unsigned lds_off, const void* gaddr,
                                            unsigned k_elems, unsigned rows,
                                            unsigned ld_elems)
{
    const unsigned long long ga = (unsigned long long)gaddr;
    u32x4 g0;
    g0[0] = 1u;                                            // count=1, user mode
    g0[1] = lds_off;                                       // lds_addr (bytes)
    g0[2] = (unsigned)(ga & 0xffffffffu);                  // global_addr[31:0]
    g0[3] = (unsigned)((ga >> 32) & 0x01ffffffu)           // global_addr[56:32]
          | (2u << 30);                                    // type=2 ("image")
    i32x8 g1;
    g1[0] = (int)(1u << 16);                               // data_size=1 -> 2B
    g1[1] = (int)((k_elems & 0xffffu) << 16);              // tensor_dim0[15:0]
    g1[2] = (int)((k_elems >> 16) | ((rows & 0xffffu) << 16)); // td0 hi | td1 lo
    g1[3] = (int)((rows >> 16) | ((k_elems & 0xffffu) << 16)); // td1 hi | tile_dim0
    g1[4] = (int)(rows & 0xffffu);                         // tile_dim1 | tile_dim2=0
    g1[5] = (int)ld_elems;                                 // tensor_dim0_stride lo
    g1[6] = 0;                                             // stride hi | td1_stride lo
    g1[7] = 0;
    const i32x4 z4 = {0, 0, 0, 0};
    const i32x8 z8 = {0, 0, 0, 0, 0, 0, 0, 0};
    __builtin_amdgcn_tensor_load_to_lds(g0, g1, z4, z4, z8, 0);
}

// A-fragment (16x32 bf16) from a row-major LDS strip, row stride ld.
// ISA layout: lanes 0-15 -> M=lane, k-halves 0..7/16..23; lanes 16-31 +8.
__device__ __forceinline__ bf16x16 load_a_frag_lds(const bf16_t* As, int ld, int k0)
{
    const int lane = threadIdx.x & 31;
    const int m  = lane & 15;
    const int kb = k0 + ((lane >> 4) << 3);
    const bf16_t* p = As + m * ld + kb;
    Frag f;
    f.q[0] = *reinterpret_cast<const Bytes16*>(p);
    f.q[1] = *reinterpret_cast<const Bytes16*>(p + 16);
    return f.v;
}

// A-fragment straight from global (used by the small-K score GEMMs).
__device__ __forceinline__ bf16x16 load_a_frag(const bf16_t* __restrict__ X,
                                               int ld, int m0, int k0)
{
    const int lane = threadIdx.x & 31;
    const int m  = m0 + (lane & 15);
    const int kb = k0 + ((lane >> 4) << 3);
    const bf16_t* p = X + (long)m * ld + kb;
    Frag f;
    f.q[0] = *reinterpret_cast<const Bytes16*>(p);
    f.q[1] = *reinterpret_cast<const Bytes16*>(p + 16);
    return f.v;
}

// B-fragment: 32x16 tile where B[k][n] = W[n][k], W row-major [N,K].
__device__ __forceinline__ bf16x16 load_b_frag(const bf16_t* __restrict__ W,
                                               int ld, int n0, int k0)
{
    const int lane = threadIdx.x & 31;
    const int n  = n0 + (lane & 15);
    const int kb = k0 + ((lane >> 4) << 3);
    const bf16_t* p = W + (long)n * ld + kb;
    Frag f;
    f.q[0] = *reinterpret_cast<const Bytes16*>(p);
    f.q[1] = *reinterpret_cast<const Bytes16*>(p + 16);
    return f.v;
}

__device__ __forceinline__ f32x8 wmma_bf16(bf16x16 a, bf16x16 b, f32x8 c)
{
    return __builtin_amdgcn_wmma_f32_16x16x32_bf16(
        false, a, false, b, (short)0, c, false, false);
}

// ---------------------------------------------------------------------------
// Projection GEMM: Y = (X @ W^T + bias) * scale, unbatched, K == 512.
// Block: 8 waves share one 16-row M-strip (TDM-staged in LDS), each wave owns
// one 16x16 N-tile. Output scatter idx = b*sb + h*sh + s*ss + d*sd.
// ---------------------------------------------------------------------------
template <int F32OUT>
__global__ __launch_bounds__(256) void proj_gemm(
    const bf16_t* __restrict__ X, int ldx,
    const bf16_t* __restrict__ W, int ldw,
    const float* __restrict__ bias,
    void* __restrict__ out,
    int S, int head_dim,
    long sb, long sh, long ss, long sd,
    int N, int K, float scale)
{
    __shared__ bf16_t As[16 * 512];

    const int wave = threadIdx.x >> 5;
    const int n0 = (blockIdx.x * 8 + wave) * 16;
    const int m0 = blockIdx.y * 16;

    if (wave == 0) {
        tdm_load_2d((unsigned)(size_t)As, X + (long)m0 * ldx,
                    (unsigned)K, 16u, (unsigned)ldx);
        __builtin_amdgcn_s_wait_tensorcnt(0);
    }
    __syncthreads();

    if (n0 < N) {
        f32x8 acc = {0.f, 0.f, 0.f, 0.f, 0.f, 0.f, 0.f, 0.f};
        for (int k0 = 0; k0 < K; k0 += 32) {
            if (k0 + 32 < K)
                __builtin_prefetch(W + (long)n0 * ldw + k0 + 32, 0, 1);
            bf16x16 a = load_a_frag_lds(As, K, k0);
            bf16x16 b = load_b_frag(W, ldw, n0, k0);
            acc = wmma_bf16(a, b, acc);
        }

        const int lane = threadIdx.x & 31;
        const int col  = lane & 15;
        const int hi   = lane >> 4;
        const int n    = n0 + col;
        const int h    = n / head_dim;
        const int d    = n - h * head_dim;
        const float bn = bias ? bias[n] : 0.0f;

#pragma unroll
        for (int r = 0; r < 8; ++r) {
            const int m = m0 + hi * 8 + r;
            const int b = m / S;
            const int s = m - b * S;
            const float y = (acc[r] + bn) * scale;
            const long idx = (long)b * sb + (long)h * sh + (long)s * ss + (long)d * sd;
            if (F32OUT) ((float*)out)[idx]  = y;
            else        ((bf16_t*)out)[idx] = (bf16_t)y;
        }
    }
}

// ---------------------------------------------------------------------------
// Batched score GEMM (small K = 64/128): Y[z] = X[z] @ W[z]^T, f32 output.
// ---------------------------------------------------------------------------
__global__ __launch_bounds__(256) void score_gemm(
    const bf16_t* __restrict__ X, long xBatch, int ldx,
    const bf16_t* __restrict__ W, long wBatch, int ldw,
    float* __restrict__ Y, long yBatch, int ldy,
    int N, int K)
{
    const int z = blockIdx.z;
    const int wave = threadIdx.x >> 5;
    const int n0 = (blockIdx.x * 8 + wave) * 16;
    const int m0 = blockIdx.y * 16;
    if (n0 >= N) return;

    const bf16_t* Xb = X + (long)z * xBatch;
    const bf16_t* Wb = W + (long)z * wBatch;

    f32x8 acc = {0.f, 0.f, 0.f, 0.f, 0.f, 0.f, 0.f, 0.f};
    for (int k0 = 0; k0 < K; k0 += 32) {
        bf16x16 a = load_a_frag(Xb, ldx, m0, k0);
        bf16x16 b = load_b_frag(Wb, ldw, n0, k0);
        acc = wmma_bf16(a, b, acc);
    }

    const int lane = threadIdx.x & 31;
    const int col  = lane & 15;
    const int hi   = lane >> 4;
    float* base = Y + (long)z * yBatch + (long)(m0 + hi * 8) * ldy + (n0 + col);
#pragma unroll
    for (int r = 0; r < 8; ++r)
        base[(long)r * ldy] = acc[r];
}

// ---------------------------------------------------------------------------
// Batched attn@V GEMM (K = 512, TDM-staged attn strip): z = b*HH + h,
// Vt per batch is [hd, S]; bf16 strided output merges heads into [B,S,D].
// ---------------------------------------------------------------------------
__global__ __launch_bounds__(256) void attn_gemm(
    const bf16_t* __restrict__ A, long aBatch, int lda,
    const bf16_t* __restrict__ Wt, long wBatch, int ldw,
    bf16_t* __restrict__ out, int HH,
    long sb, long sh, long ss, long sd,
    int N, int K)
{
    __shared__ bf16_t As[16 * 512];

    const int z = blockIdx.z;
    const int wave = threadIdx.x >> 5;
    const int n0 = (blockIdx.x * 8 + wave) * 16;
    const int m0 = blockIdx.y * 16;

    const bf16_t* Ab = A  + (long)z * aBatch;
    const bf16_t* Wb = Wt + (long)z * wBatch;

    if (wave == 0) {
        tdm_load_2d((unsigned)(size_t)As, Ab + (long)m0 * lda,
                    (unsigned)K, 16u, (unsigned)lda);
        __builtin_amdgcn_s_wait_tensorcnt(0);
    }
    __syncthreads();

    if (n0 < N) {
        f32x8 acc = {0.f, 0.f, 0.f, 0.f, 0.f, 0.f, 0.f, 0.f};
        for (int k0 = 0; k0 < K; k0 += 32) {
            if (k0 + 32 < K)
                __builtin_prefetch(Wb + (long)n0 * ldw + k0 + 32, 0, 1);
            bf16x16 a = load_a_frag_lds(As, K, k0);
            bf16x16 b = load_b_frag(Wb, ldw, n0, k0);
            acc = wmma_bf16(a, b, acc);
        }

        const int lane = threadIdx.x & 31;
        const int col  = lane & 15;
        const int hi   = lane >> 4;
        const int b = z / HH;
        const int h = z - b * HH;
        const long base = (long)b * sb + (long)h * sh + (long)(n0 + col) * sd;
#pragma unroll
        for (int r = 0; r < 8; ++r) {
            const int m = m0 + hi * 8 + r;
            out[base + (long)m * ss] = (bf16_t)acc[r];
        }
    }
}

// ---------------------------------------------------------------------------
// Row softmax with optional relative-position bias gather.
// bias(l,r) = P[l, clip(r-l,-256,256)+256]
// ---------------------------------------------------------------------------
__global__ __launch_bounds__(256) void softmax_rows(
    const float* __restrict__ scores,
    const float* __restrict__ P, int Pld,
    bf16_t* __restrict__ attn, int S)
{
    const int z = blockIdx.y;
    const int l = blockIdx.x;
    const float* srow = scores + ((long)z * S + l) * S;
    bf16_t* arow = attn + ((long)z * S + l) * S;
    const int tid = threadIdx.x;
    const int r0 = tid, r1 = tid + 256;

    float v0 = srow[r0];
    float v1 = srow[r1];
    if (P) {
        const float* prow = P + ((long)z * S + l) * Pld;
        int m0 = min(256, max(-256, r0 - l)) + 256;
        int m1 = min(256, max(-256, r1 - l)) + 256;
        v0 += prow[m0];
        v1 += prow[m1];
    }

    __shared__ float red[256];
    red[tid] = fmaxf(v0, v1);
    __syncthreads();
    for (int s = 128; s > 0; s >>= 1) {
        if (tid < s) red[tid] = fmaxf(red[tid], red[tid + s]);
        __syncthreads();
    }
    const float mx = red[0];
    __syncthreads();

    const float e0 = __expf(v0 - mx);
    const float e1 = __expf(v1 - mx);
    red[tid] = e0 + e1;
    __syncthreads();
    for (int s = 128; s > 0; s >>= 1) {
        if (tid < s) red[tid] += red[tid + s];
        __syncthreads();
    }
    const float inv = 1.0f / red[0];
    arow[r0] = (bf16_t)(e0 * inv);
    arow[r1] = (bf16_t)(e1 * inv);
}

// ---------------------------------------------------------------------------
// Elementwise helpers
// ---------------------------------------------------------------------------
__global__ __launch_bounds__(256) void cvt_f32_bf16(
    const float* __restrict__ in, bf16_t* __restrict__ out, long n)
{
    long i = (long)blockIdx.x * blockDim.x + threadIdx.x;
    const long stride = (long)gridDim.x * blockDim.x;
    for (; i < n; i += stride) out[i] = (bf16_t)in[i];
}

__global__ __launch_bounds__(256) void combine_kernel(
    const bf16_t* __restrict__ a, const bf16_t* __restrict__ b,
    bf16_t* __restrict__ out, long n)
{
    long i = (long)blockIdx.x * blockDim.x + threadIdx.x;
    const long stride = (long)gridDim.x * blockDim.x;
    for (; i < n; i += stride)
        out[i] = (bf16_t)(0.7f * (float)a[i] + 0.3f * (float)b[i]);
}

// ---------------------------------------------------------------------------
// Host orchestration
// ---------------------------------------------------------------------------
extern "C" void kernel_launch(void* const* d_in, const int* in_sizes, int n_in,
                              void* d_out, int out_size, void* d_ws, size_t ws_size,
                              hipStream_t stream)
{
    (void)in_sizes; (void)n_in; (void)out_size; (void)ws_size;

    const int B = 32, S = 512, D = 512, H = 8, HD = 64, HG = 4, HDG = 128;
    const long M = (long)B * S;              // 16384
    const int RELROWS = 513, RELPAD = 528;   // N padded to /16

    const float* query  = (const float*)d_in[0];
    const float* key    = (const float*)d_in[1];
    const float* value  = (const float*)d_in[2];
    const float* wq     = (const float*)d_in[3];
    const float* bq     = (const float*)d_in[4];
    const float* wk     = (const float*)d_in[5];
    const float* bk     = (const float*)d_in[6];
    const float* wv     = (const float*)d_in[7];
    const float* bv     = (const float*)d_in[8];
    const float* wo     = (const float*)d_in[9];
    const float* bo     = (const float*)d_in[10];
    const float* rel_k  = (const float*)d_in[11];
    const float* g_in_w = (const float*)d_in[12];
    const float* g_in_b = (const float*)d_in[13];
    const float* g_out_w= (const float*)d_in[14];
    const float* g_out_b= (const float*)d_in[15];

    char* ws = (char*)d_ws;
    size_t off = 0;
    auto alloc = [&](size_t bytes) -> void* {
        void* p = ws + off;
        off += (bytes + 255) & ~(size_t)255;
        return p;
    };

    bf16_t* Xq  = (bf16_t*)alloc(M * D * 2);
    bf16_t* Xk  = (bf16_t*)alloc(M * D * 2);
    bf16_t* Xv  = (bf16_t*)alloc(M * D * 2);
    bf16_t* Wq  = (bf16_t*)alloc((long)D * D * 2);
    bf16_t* Wk  = (bf16_t*)alloc((long)D * D * 2);
    bf16_t* Wv  = (bf16_t*)alloc((long)D * D * 2);
    bf16_t* Wo  = (bf16_t*)alloc((long)D * D * 2);
    bf16_t* Wgi = (bf16_t*)alloc((long)3 * D * D * 2);
    bf16_t* Wgo = (bf16_t*)alloc((long)D * D * 2);
    bf16_t* Rel = (bf16_t*)alloc((long)RELPAD * HD * 2);

    bf16_t* qb  = (bf16_t*)alloc(M * D * 2);    // [B,H,S,64]
    bf16_t* kb  = (bf16_t*)alloc(M * D * 2);    // [B,H,S,64], pre-scaled 1/8
    bf16_t* vT  = (bf16_t*)alloc(M * D * 2);    // [B,H,64,S]
    bf16_t* qgb = (bf16_t*)alloc(M * D * 2);    // [B,4,S,128]
    bf16_t* kgb = (bf16_t*)alloc(M * D * 2);    // [B,4,S,128], pre-scaled
    bf16_t* vgT = (bf16_t*)alloc(M * D * 2);    // [B,4,128,S]

    float*  scL = (float*) alloc((long)B * H  * S * S * 4);
    float*  Pbf = (float*) alloc((long)B * H  * S * RELPAD * 4);
    bf16_t* atL = (bf16_t*)alloc((long)B * H  * S * S * 2);
    float*  scG = (float*) alloc((long)B * HG * S * S * 4);
    bf16_t* atG = (bf16_t*)alloc((long)B * HG * S * S * 2);

    bf16_t* ctxL = (bf16_t*)alloc(M * D * 2);
    bf16_t* ctxG = (bf16_t*)alloc(M * D * 2);
    bf16_t* Gout = (bf16_t*)alloc(M * D * 2);
    bf16_t* comb = (bf16_t*)alloc(M * D * 2);

    // ---- stage 0: casts (rel table zero-padded to 528 rows)
    (void)hipMemsetAsync(Rel, 0, (long)RELPAD * HD * 2, stream);
    {
        dim3 blk(256);
        auto g = [](long n) { long b = (n + 255) / 256; return dim3((unsigned)(b > 4096 ? 4096 : b)); };
        cvt_f32_bf16<<<g(M * D), blk, 0, stream>>>(query,  Xq,  M * D);
        cvt_f32_bf16<<<g(M * D), blk, 0, stream>>>(key,    Xk,  M * D);
        cvt_f32_bf16<<<g(M * D), blk, 0, stream>>>(value,  Xv,  M * D);
        cvt_f32_bf16<<<g((long)D * D), blk, 0, stream>>>(wq, Wq, (long)D * D);
        cvt_f32_bf16<<<g((long)D * D), blk, 0, stream>>>(wk, Wk, (long)D * D);
        cvt_f32_bf16<<<g((long)D * D), blk, 0, stream>>>(wv, Wv, (long)D * D);
        cvt_f32_bf16<<<g((long)D * D), blk, 0, stream>>>(wo, Wo, (long)D * D);
        cvt_f32_bf16<<<g((long)3 * D * D), blk, 0, stream>>>(g_in_w, Wgi, (long)3 * D * D);
        cvt_f32_bf16<<<g((long)D * D), blk, 0, stream>>>(g_out_w, Wgo, (long)D * D);
        cvt_f32_bf16<<<g((long)RELROWS * HD), blk, 0, stream>>>(rel_k, Rel, (long)RELROWS * HD);
    }

    const dim3 blk(256);
    const dim3 gProj(4, (unsigned)(M / 16));
    const float kScale  = 0.125f;               // 1/sqrt(64)
    const float kgScale = 0.08838834764831845f; // 1/sqrt(128)

    // ---- stage 1: projections (scale folded into K / Kg)
    proj_gemm<0><<<gProj, blk, 0, stream>>>(Xq, D, Wq, D, bq, qb, S, HD,
        (long)H*S*HD, (long)S*HD, (long)HD, 1L, D, D, 1.0f);
    proj_gemm<0><<<gProj, blk, 0, stream>>>(Xk, D, Wk, D, bk, kb, S, HD,
        (long)H*S*HD, (long)S*HD, (long)HD, 1L, D, D, kScale);
    proj_gemm<0><<<gProj, blk, 0, stream>>>(Xv, D, Wv, D, bv, vT, S, HD,
        (long)H*HD*S, (long)HD*S, 1L, (long)S, D, D, 1.0f);           // transposed V

    proj_gemm<0><<<gProj, blk, 0, stream>>>(Xq, D, Wgi, D, g_in_b, qgb, S, HDG,
        (long)HG*S*HDG, (long)S*HDG, (long)HDG, 1L, D, D, 1.0f);
    proj_gemm<0><<<gProj, blk, 0, stream>>>(Xk, D, Wgi + (long)D*D, D, g_in_b + D, kgb, S, HDG,
        (long)HG*S*HDG, (long)S*HDG, (long)HDG, 1L, D, D, kgScale);
    proj_gemm<0><<<gProj, blk, 0, stream>>>(Xv, D, Wgi + (long)2*D*D, D, g_in_b + 2*D, vgT, S, HDG,
        (long)HG*HDG*S, (long)HDG*S, 1L, (long)S, D, D, 1.0f);        // transposed Vg

    // ---- stage 2: attention scores (batched WMMA GEMMs)
    score_gemm<<<dim3(4, 32, B * H), blk, 0, stream>>>(
        qb, (long)S*HD, HD, kb, (long)S*HD, HD,
        scL, (long)S*S, S, S, HD);
    score_gemm<<<dim3(5, 32, B * H), blk, 0, stream>>>(
        qb, (long)S*HD, HD, Rel, 0L, HD,
        Pbf, (long)S*RELPAD, RELPAD, RELPAD, HD);
    score_gemm<<<dim3(4, 32, B * HG), blk, 0, stream>>>(
        qgb, (long)S*HDG, HDG, kgb, (long)S*HDG, HDG,
        scG, (long)S*S, S, S, HDG);

    // ---- stage 3: softmax (+ rel bias gather for local)
    softmax_rows<<<dim3(S, B * H),  blk, 0, stream>>>(scL, Pbf, RELPAD, atL, S);
    softmax_rows<<<dim3(S, B * HG), blk, 0, stream>>>(scG, nullptr, 0, atG, S);

    // ---- stage 4: attn @ V (heads merged back to [B,S,D] in epilogue)
    attn_gemm<<<dim3(1, 32, B * H), blk, 0, stream>>>(
        atL, (long)S*S, S, vT, (long)HD*S, S,
        ctxL, H, (long)S*D, (long)HD, (long)D, 1L, HD, S);
    attn_gemm<<<dim3(1, 32, B * HG), blk, 0, stream>>>(
        atG, (long)S*S, S, vgT, (long)HDG*S, S,
        ctxG, HG, (long)S*D, (long)HDG, (long)D, 1L, HDG, S);

    // ---- stage 5: global out-proj, combine, final projection (f32 -> d_out)
    proj_gemm<0><<<gProj, blk, 0, stream>>>(ctxG, D, Wgo, D, g_out_b, Gout, S, D,
        (long)S*D, 0L, (long)D, 1L, D, D, 1.0f);

    combine_kernel<<<dim3(4096), blk, 0, stream>>>(ctxL, Gout, comb, M * D);

    proj_gemm<1><<<gProj, blk, 0, stream>>>(comb, D, Wo, D, bo, d_out, S, D,
        (long)S*D, 0L, (long)D, 1L, D, D, 1.0f);
}
